// GPT_21234318312244
// MI455X (gfx1250) — matrix operand order repaired and checked by reading, not compiled, not run on hardware
//
#include <hip/hip_runtime.h>

// ---------------- problem constants ----------------
#define BB 8
#define TT 1024
#define VV 8192
#define DD 1024
#define HH 16
#define HD 64
#define LL 6
#define BT (BB*TT)          // 8192
#define D3 (3*DD)           // 3072

// ---------------- types ----------------
typedef __attribute__((ext_vector_type(16))) __bf16 v16bf;
typedef __attribute__((ext_vector_type(8)))  float  v8f;
typedef __attribute__((ext_vector_type(4)))  unsigned int u32x4;

union Frag16 { v16bf v; u32x4 q[2]; };

static __device__ __forceinline__ __bf16 tobf(float f) {
    unsigned u = __builtin_bit_cast(unsigned, f);
    unsigned r = u + 0x7FFFu + ((u >> 16) & 1u);   // round-to-nearest-even
    unsigned short h = (unsigned short)(r >> 16);
    return __builtin_bit_cast(__bf16, h);
}

// ---------------- token recovery: argwhere(one_hot) ----------------
__global__ __launch_bounds__(256) void tok_kernel(const float* __restrict__ idx,
                                                  int* __restrict__ tok) {
    const int row = blockIdx.x;                 // 0..BT-1
    const float* p = idx + (size_t)row * VV;
    int t = threadIdx.x, found = 0;
    for (int i = t; i < VV; i += 256) if (p[i] > 0.5f) found = i;
    __shared__ int red[256];
    red[t] = found; __syncthreads();
    for (int s = 128; s > 0; s >>= 1) { if (t < s) red[t] = max(red[t], red[t + s]); __syncthreads(); }
    if (t == 0) tok[row] = red[0];
}

// ---------------- embedding gather: x[bt,d] = W_emb[d, tok] ----------------
__global__ __launch_bounds__(256) void embed_kernel(const float* __restrict__ Wemb,
                                                    const int* __restrict__ tok,
                                                    float* __restrict__ x) {
    const int row = blockIdx.x;
    const int tk = tok[row];
    float* xp = x + (size_t)row * DD;
    for (int d = threadIdx.x; d < DD; d += 256)
        xp[d] = Wemb[(size_t)d * VV + tk];
}

// ---------------- f32 -> bf16 convert ----------------
__global__ __launch_bounds__(256) void cvt_kernel(const float* __restrict__ in,
                                                  __bf16* __restrict__ out, int n) {
    int i = blockIdx.x * 256 + threadIdx.x;
    if (i < n) out[i] = tobf(in[i]);
}

// ---------------- parameterless LayerNorm, bf16 output ----------------
__global__ __launch_bounds__(256) void ln_kernel(const float* __restrict__ x,
                                                 __bf16* __restrict__ out) {
    const int row = blockIdx.x;
    const float* p = x + (size_t)row * DD;
    const int t = threadIdx.x;
    float v[4], s = 0.f, sq = 0.f;
    #pragma unroll
    for (int i = 0; i < 4; ++i) { v[i] = p[t + i * 256]; s += v[i]; sq += v[i] * v[i]; }
    __shared__ float rs[256], rq[256];
    rs[t] = s; rq[t] = sq; __syncthreads();
    for (int k = 128; k > 0; k >>= 1) {
        if (t < k) { rs[t] += rs[t + k]; rq[t] += rq[t + k]; }
        __syncthreads();
    }
    const float mean = rs[0] * (1.0f / DD);
    const float var  = rq[0] * (1.0f / DD) - mean * mean;
    const float rstd = rsqrtf(var + 1e-5f);
    __bf16* op = out + (size_t)row * DD;
    #pragma unroll
    for (int i = 0; i < 4; ++i) op[t + i * 256] = tobf((v[i] - mean) * rstd);
}

// ---------------- bf16 NT GEMM: C[M,N] = A[M,K] * B[N,K]^T ----------------
// 256 threads = 8 waves; block tile 256x64; K-step 32; wave tile 32x64
// (2 A frags x 4 B frags -> 8 WMMAs per K-step, 1.5 ds_load_b128 per WMMA)
__global__ __launch_bounds__(256) void gemm_bf16(const __bf16* __restrict__ A,
                                                 const __bf16* __restrict__ Bw,
                                                 void* __restrict__ Cout,
                                                 int M, int N, int K, int out_bf16) {
    __shared__ __attribute__((aligned(16))) __bf16 At[256][40];
    __shared__ __attribute__((aligned(16))) __bf16 Bt[64][40];
    const int t = threadIdx.x;
    const int w = t >> 5, lane = t & 31;
    const int n0 = lane & 15, hi = lane >> 4;
    const int rowBase = blockIdx.y * 256;
    const int colBase = blockIdx.x * 64;

    v8f acc[2][4];
    #pragma unroll
    for (int m = 0; m < 2; ++m)
        #pragma unroll
        for (int j = 0; j < 4; ++j)
            #pragma unroll
            for (int i = 0; i < 8; ++i) acc[m][j][i] = 0.f;

    for (int k0 = 0; k0 < K; k0 += 32) {
        // prefetch next K-tile (lowers to global_prefetch_b8)
        if (k0 + 32 < K) {
            __builtin_prefetch(A + (size_t)(rowBase + (t >> 2)) * K + k0 + 32 + (t & 3) * 8, 0, 0);
            __builtin_prefetch(Bw + (size_t)(colBase + (t >> 2)) * K + k0 + 32 + (t & 3) * 8, 0, 0);
        }
        #pragma unroll
        for (int i = 0; i < 4; ++i) {           // A tile: 256 rows x 32 k (1024 x 16B chunks)
            int c = t + i * 256, r = c >> 2, kc = c & 3;
            *(u32x4*)(&At[r][kc * 8]) =
                *(const u32x4*)(A + (size_t)(rowBase + r) * K + k0 + kc * 8);
        }
        {                                        // B tile: 64 rows x 32 k
            int r = t >> 2, kc = t & 3;
            *(u32x4*)(&Bt[r][kc * 8]) =
                *(const u32x4*)(Bw + (size_t)(colBase + r) * K + k0 + kc * 8);
        }
        __syncthreads();

        Frag16 a[2];                             // A frags: m=n0; k = hi*8+{0..7}, 16+hi*8+{0..7}
        #pragma unroll
        for (int m = 0; m < 2; ++m) {
            a[m].q[0] = *(const u32x4*)(&At[32 * w + 16 * m + n0][hi * 8]);
            a[m].q[1] = *(const u32x4*)(&At[32 * w + 16 * m + n0][16 + hi * 8]);
        }
        #pragma unroll
        for (int j = 0; j < 4; ++j) {            // B frag: n=n0; k = hi*8+{0..7}, 16+hi*8+{0..7}
            Frag16 b;
            b.q[0] = *(const u32x4*)(&Bt[16 * j + n0][hi * 8]);
            b.q[1] = *(const u32x4*)(&Bt[16 * j + n0][16 + hi * 8]);
            #pragma unroll
            for (int m = 0; m < 2; ++m)
                acc[m][j] = __builtin_amdgcn_wmma_f32_16x16x32_bf16(false, a[m].v, false, b.v,
                                                                    (short)0, acc[m][j], false, false);
        }
        __syncthreads();
    }

    if (out_bf16) {
        __bf16* Cb = (__bf16*)Cout;
        #pragma unroll
        for (int m = 0; m < 2; ++m)
            #pragma unroll
            for (int j = 0; j < 4; ++j)
                #pragma unroll
                for (int r = 0; r < 8; ++r)
                    Cb[(size_t)(rowBase + 32 * w + 16 * m + r + 8 * hi) * N +
                       colBase + 16 * j + n0] = tobf(acc[m][j][r]);
    } else {
        float* Cf = (float*)Cout;
        #pragma unroll
        for (int m = 0; m < 2; ++m)
            #pragma unroll
            for (int j = 0; j < 4; ++j)
                #pragma unroll
                for (int r = 0; r < 8; ++r)
                    Cf[(size_t)(rowBase + 32 * w + 16 * m + r + 8 * hi) * N +
                       colBase + 16 * j + n0] = acc[m][j][r];
    }
}

// ---------------- flash attention, adds head output into residual x ----------------
// grid (T/128, H, B); 256 threads = 8 waves; wave owns 16 q rows; key blocks of 32
__global__ __launch_bounds__(256) void flash_kernel(const __bf16* __restrict__ qkv,
                                                    float* __restrict__ x) {
    __shared__ __attribute__((aligned(16))) __bf16 Kt[32][72];       // [key][d]
    __shared__ __attribute__((aligned(16))) __bf16 Vt[64][40];       // [d][key] (transposed)
    __shared__ __attribute__((aligned(16))) __bf16 Pw[8][16][40];    // per-wave P staging

    const int qblk = blockIdx.x, h = blockIdx.y, b = blockIdx.z;
    const int t = threadIdx.x;
    const int w = t >> 5, lane = t & 31;
    const int n0 = lane & 15, hi = lane >> 4;
    const int q0 = qblk * 128;
    const int qw0 = q0 + 16 * w;

    // Q fragments (A layout), kept in registers for the whole kernel
    Frag16 qf[2];
    {
        const __bf16* qp = qkv + ((size_t)b * TT + qw0 + n0) * D3 + h * HD;
        #pragma unroll
        for (int s = 0; s < 2; ++s) {
            qf[s].q[0] = *(const u32x4*)(qp + s * 32 + hi * 8);
            qf[s].q[1] = *(const u32x4*)(qp + s * 32 + 16 + hi * 8);
        }
    }

    float mrow[8], lrow[8];
    v8f Oacc[4];
    #pragma unroll
    for (int r = 0; r < 8; ++r) { mrow[r] = -3.0e38f; lrow[r] = 0.f; }
    #pragma unroll
    for (int j = 0; j < 4; ++j)
        #pragma unroll
        for (int i = 0; i < 8; ++i) Oacc[j][i] = 0.f;

    const int nkb = (q0 + 128) / 32;
    for (int kb = 0; kb < nkb; ++kb) {
        const int kk0 = kb * 32;
        {   // K tile: thread -> (key = t/8, dchunk = t%8)
            int key = t >> 3, dc = t & 7;
            *(u32x4*)(&Kt[key][dc * 8]) =
                *(const u32x4*)(qkv + ((size_t)b * TT + kk0 + key) * D3 + DD + h * HD + dc * 8);
        }
        {   // V tile transposed: thread -> (key = t%32, dbase = (t/32)*8)
            int key = t & 31, db = (t >> 5) * 8;
            u32x4 vv = *(const u32x4*)(qkv + ((size_t)b * TT + kk0 + key) * D3 + 2 * DD + h * HD + db);
            const __bf16* ve = (const __bf16*)&vv;
            #pragma unroll
            for (int i = 0; i < 8; ++i) Vt[db + i][key] = ve[i];
        }
        __syncthreads();

        if (kk0 <= qw0 + 15) {
            // S = (Q K^T) / sqrt(HD); two 16x16 key tiles
            v8f S[2];
            #pragma unroll
            for (int j = 0; j < 2; ++j)
                #pragma unroll
                for (int i = 0; i < 8; ++i) S[j][i] = 0.f;
            #pragma unroll
            for (int s = 0; s < 2; ++s) {
                #pragma unroll
                for (int j = 0; j < 2; ++j) {
                    Frag16 bk;
                    bk.q[0] = *(const u32x4*)(&Kt[16 * j + n0][s * 32 + hi * 8]);
                    bk.q[1] = *(const u32x4*)(&Kt[16 * j + n0][s * 32 + 16 + hi * 8]);
                    S[j] = __builtin_amdgcn_wmma_f32_16x16x32_bf16(false, qf[s].v, false, bk.v,
                                                                   (short)0, S[j], false, false);
                }
            }
            // online softmax (row r+8*hi per lane; 16-lane shfl reductions)
            #pragma unroll
            for (int r = 0; r < 8; ++r) {
                const int row = qw0 + r + 8 * hi;
                float s0 = S[0][r] * 0.125f;
                float s1 = S[1][r] * 0.125f;
                if (kk0 + n0 > row)      s0 = -3.0e38f;
                if (kk0 + 16 + n0 > row) s1 = -3.0e38f;
                float mx = fmaxf(s0, s1);
                #pragma unroll
                for (int msk = 8; msk >= 1; msk >>= 1) mx = fmaxf(mx, __shfl_xor(mx, msk, 32));
                const float newm = fmaxf(mrow[r], mx);
                const float alpha = __expf(mrow[r] - newm);
                const float p0 = __expf(s0 - newm);
                const float p1 = __expf(s1 - newm);
                float rs = p0 + p1;
                #pragma unroll
                for (int msk = 8; msk >= 1; msk >>= 1) rs += __shfl_xor(rs, msk, 32);
                lrow[r] = lrow[r] * alpha + rs;
                mrow[r] = newm;
                #pragma unroll
                for (int j = 0; j < 4; ++j) Oacc[j][r] *= alpha;
                Pw[w][r + 8 * hi][n0]      = tobf(p0);
                Pw[w][r + 8 * hi][16 + n0] = tobf(p1);
            }
            // O += P @ V  (contraction over 32 keys)
            Frag16 pa;
            pa.q[0] = *(const u32x4*)(&Pw[w][n0][hi * 8]);
            pa.q[1] = *(const u32x4*)(&Pw[w][n0][16 + hi * 8]);
            #pragma unroll
            for (int j = 0; j < 4; ++j) {
                Frag16 bv;
                bv.q[0] = *(const u32x4*)(&Vt[16 * j + n0][hi * 8]);
                bv.q[1] = *(const u32x4*)(&Vt[16 * j + n0][16 + hi * 8]);
                Oacc[j] = __builtin_amdgcn_wmma_f32_16x16x32_bf16(false, pa.v, false, bv.v,
                                                                  (short)0, Oacc[j], false, false);
            }
        }
        __syncthreads();
    }

    // epilogue: x[row, h*64 + d] += O / l   (disjoint per-head columns; no atomics needed)
    #pragma unroll
    for (int r = 0; r < 8; ++r) {
        const float inv = 1.0f / lrow[r];
        const int row = qw0 + r + 8 * hi;
        float* xp = x + ((size_t)b * TT + row) * DD + h * HD;
        #pragma unroll
        for (int j = 0; j < 4; ++j) xp[16 * j + n0] += Oacc[j][r] * inv;
    }
}

// ---------------- per-row NLL over V ----------------
__global__ __launch_bounds__(256) void loss_rows(const float* __restrict__ logits,
                                                 const int* __restrict__ targets,
                                                 float* __restrict__ rownll) {
    const int row = blockIdx.x;
    const float* p = logits + (size_t)row * VV;
    const int t = threadIdx.x;
    __shared__ float red[256];
    float mx = -3.0e38f;
    for (int i = t; i < VV; i += 256) mx = fmaxf(mx, p[i]);
    red[t] = mx; __syncthreads();
    for (int k = 128; k > 0; k >>= 1) { if (t < k) red[t] = fmaxf(red[t], red[t + k]); __syncthreads(); }
    const float M = red[0];
    __syncthreads();
    float s = 0.f;
    for (int i = t; i < VV; i += 256) s += __expf(p[i] - M);
    red[t] = s; __syncthreads();
    for (int k = 128; k > 0; k >>= 1) { if (t < k) red[t] += red[t + k]; __syncthreads(); }
    if (t == 0) rownll[row] = M + __logf(red[0]) - p[targets[row]];
}

__global__ __launch_bounds__(256) void loss_final(const float* __restrict__ rownll,
                                                  float* __restrict__ out) {
    const int t = threadIdx.x;
    __shared__ float red[256];
    float s = 0.f;
    for (int i = t; i < BT; i += 256) s += rownll[i];
    red[t] = s; __syncthreads();
    for (int k = 128; k > 0; k >>= 1) { if (t < k) red[t] += red[t + k]; __syncthreads(); }
    if (t == 0) out[0] = red[0] * (1.0f / BT);
}

// ---------------- host orchestration ----------------
extern "C" void kernel_launch(void* const* d_in, const int* in_sizes, int n_in,
                              void* d_out, int out_size, void* d_ws, size_t ws_size,
                              hipStream_t stream) {
    const float* idx     = (const float*)d_in[0];   // [B,T,V] one-hot f32
    const int*   targets = (const int*)  d_in[1];   // [B,T]
    const float* Wemb    = (const float*)d_in[2];   // [D,V]
    const float* Wattn   = (const float*)d_in[3];   // [L,3D,D]
    const float* Wout    = (const float*)d_in[4];   // [V,D]
    float* out = (float*)d_out;                     // logits [B,T,V] ++ loss [1]

    char* ws = (char*)d_ws;
    size_t off = 0;
    int*    tok   = (int*)   (ws + off); off += (size_t)BT * 4;                 // 32 KB
    float*  x     = (float*) (ws + off); off += (size_t)BT * DD * 4;            // 32 MB
    __bf16* hbf   = (__bf16*)(ws + off); off += (size_t)BT * DD * 2;            // 16 MB
    __bf16* qkvb  = (__bf16*)(ws + off); off += (size_t)BT * D3 * 2;            // 48 MB
    __bf16* Wab   = (__bf16*)(ws + off); off += (size_t)LL * D3 * DD * 2;       // 36 MB
    __bf16* Wob   = (__bf16*)(ws + off); off += (size_t)VV * DD * 2;            // 16 MB
    float*  rnll  = (float*) (ws + off); off += (size_t)BT * 4;                 // 32 KB

    // 1) tokens + embedding gather (avoids the dense one-hot GEMM entirely)
    tok_kernel  <<<BT, 256, 0, stream>>>(idx, tok);
    embed_kernel<<<BT, 256, 0, stream>>>(Wemb, tok, x);

    // 2) weight precision conversion (once per call)
    {
        int nA = LL * D3 * DD;   // 18,874,368
        int nO = VV * DD;        //  8,388,608
        cvt_kernel<<<nA / 256, 256, 0, stream>>>(Wattn, Wab, nA);
        cvt_kernel<<<nO / 256, 256, 0, stream>>>(Wout,  Wob, nO);
    }

    // 3) transformer blocks
    for (int l = 0; l < LL; ++l) {
        ln_kernel<<<BT, 256, 0, stream>>>(x, hbf);
        gemm_bf16<<<dim3(D3 / 64, BT / 256), 256, 0, stream>>>(
            hbf, Wab + (size_t)l * D3 * DD, qkvb, BT, D3, DD, 1);
        flash_kernel<<<dim3(TT / 128, HH, BB), 256, 0, stream>>>(qkvb, x);
    }

    // 4) final LN + logits GEMM (f32 output straight into d_out)
    ln_kernel<<<BT, 256, 0, stream>>>(x, hbf);
    gemm_bf16<<<dim3(VV / 64, BT / 256), 256, 0, stream>>>(hbf, Wob, out, BT, VV, DD, 0);

    // 5) loss
    loss_rows <<<BT, 256, 0, stream>>>(out, targets, rnll);
    loss_final<<<1,  256, 0, stream>>>(rnll, out + (size_t)BT * VV);
}
